// WeightedEntropicOTDistanceV2_15934328668518
// MI455X (gfx1250) — matrix operand order repaired and checked by reading, not compiled, not run on hardware
//
#include <hip/hip_runtime.h>

// Problem constants (fixed by the reference setup)
#define BB 64
#define NN 512
#define MM 512
#define DD 128
#define INV_EPS 20.0f                 // 1/0.05
#define LOG_A (-6.23832462503951f)    // log(1/512)
#define LOG_B (-6.23832462503951f)

typedef _Float16 half_t;
typedef __attribute__((ext_vector_type(16))) half_t v16h;
typedef __attribute__((ext_vector_type(8)))  half_t v8h;
typedef __attribute__((ext_vector_type(8)))  float  v8f;

static __device__ __forceinline__ v16h cat16(v8h a, v8h b) {
    v16h r;
#pragma unroll
    for (int i = 0; i < 8; ++i) { r[i] = a[i]; r[i + 8] = b[i]; }
    return r;
}

// ---------------------------------------------------------------------------
// Kernel 0: per-row inverse L2 norms (1/max(||x||,1e-12)); one wave per row.
// ---------------------------------------------------------------------------
__global__ __launch_bounds__(256) void norms_kernel(const float* __restrict__ x,
                                                    float* __restrict__ inv_norm) {
    int wid = threadIdx.x >> 5, lane = threadIdx.x & 31;
    int row = blockIdx.x * 8 + wid;               // B*N rows total
    const float4 v = ((const float4*)(x + (size_t)row * DD))[lane];
    float s = v.x * v.x + v.y * v.y + v.z * v.z + v.w * v.w;
#pragma unroll
    for (int off = 16; off; off >>= 1) s += __shfl_xor(s, off, 32);
    if (lane == 0) inv_norm[row] = 1.0f / fmaxf(sqrtf(s), 1e-12f);
}

// ---------------------------------------------------------------------------
// Kernel 1: zero-fill (for log_u / log_v init)
// ---------------------------------------------------------------------------
__global__ __launch_bounds__(256) void zero_kernel(float* __restrict__ p) {
    p[blockIdx.x * 256 + threadIdx.x] = 0.0f;
}

// ---------------------------------------------------------------------------
// Kernel 2: cost matrix via WMMA.  C[b,i,j] = max(2 - 2 * shat_i . that_j, 0)
// Split-precision: f32 value = hi(f16) + lo(f16); G ~= Ahi*Bhi + Ahi*Blo + Alo*Bhi
// Workgroup = 256 threads (8 waves), computes a 64x64 tile for one batch.
// K=128 processed in two 64-wide phases to keep LDS at ~36KB.
// ---------------------------------------------------------------------------
__global__ __launch_bounds__(256) void cost_kernel(const float* __restrict__ S,
                                                   const float* __restrict__ T,
                                                   const float* __restrict__ inv_ns,
                                                   const float* __restrict__ inv_nt,
                                                   float* __restrict__ cost) {
    int bid  = blockIdx.x;
    int b    = bid >> 6;
    int tile = bid & 63;
    int i0   = (tile >> 3) * 64;
    int j0   = (tile & 7) * 64;

    __shared__ half_t sAhi[64][72];   // 64 rows x 64 K-halves (+8 pad)
    __shared__ half_t sAlo[64][72];
    __shared__ half_t sBhi[64][72];
    __shared__ half_t sBlo[64][72];

    const float* Sb  = S + ((size_t)b * NN + i0) * DD;
    const float* Tb  = T + ((size_t)b * MM + j0) * DD;
    const float* ins = inv_ns + b * NN + i0;
    const float* inT = inv_nt + b * MM + j0;

    int lane = threadIdx.x & 31, w = threadIdx.x >> 5;
    int r  = w & 3;               // A row group (16 rows)
    int cg = (w >> 2) << 1;       // two B col groups per wave
    int lg = lane >> 4;           // lane group within wave
    int li = lane & 15;

    v8f acc[2];
    acc[0] = (v8f){};
    acc[1] = (v8f){};

    for (int ph = 0; ph < 2; ++ph) {
        __syncthreads();          // previous phase compute done before overwrite
        int kOff = ph * 64;
        // --- stage tiles: 64x64 floats each, 1024 float4 per matrix ---
#pragma unroll
        for (int it = 0; it < 4; ++it) {
            int idx4 = threadIdx.x + it * 256;          // 0..1023
            int row  = idx4 >> 4;                       // 16 float4 per row
            int c4   = (idx4 & 15) << 2;
            float4 v = *(const float4*)(Sb + (size_t)row * DD + kOff + c4);
            float inv = ins[row];
            float vx[4] = {v.x * inv, v.y * inv, v.z * inv, v.w * inv};
#pragma unroll
            for (int k = 0; k < 4; ++k) {
                half_t h = (half_t)vx[k];
                sAhi[row][c4 + k] = h;
                sAlo[row][c4 + k] = (half_t)(vx[k] - (float)h);
            }
            v   = *(const float4*)(Tb + (size_t)row * DD + kOff + c4);
            inv = inT[row];
            float wx[4] = {v.x * inv, v.y * inv, v.z * inv, v.w * inv};
#pragma unroll
            for (int k = 0; k < 4; ++k) {
                half_t h = (half_t)wx[k];
                sBhi[row][c4 + k] = h;
                sBlo[row][c4 + k] = (half_t)(wx[k] - (float)h);
            }
        }
        __syncthreads();
        // --- compute: 2 K-chunks of 32, 2 col subtiles, 3 WMMAs each ---
        int m = r * 16 + li;
#pragma unroll
        for (int cc = 0; cc < 2; ++cc) {
            int n = (cg + cc) * 16 + li;
#pragma unroll
            for (int kc = 0; kc < 2; ++kc) {
                int kb = kc * 32;
                // A fragment: lane<16 -> K {kb..kb+7, kb+16..23}; lane>=16 shifted by 8
                v16h ahi = cat16(*(const v8h*)&sAhi[m][kb + lg * 8],
                                 *(const v8h*)&sAhi[m][kb + 16 + lg * 8]);
                v16h alo = cat16(*(const v8h*)&sAlo[m][kb + lg * 8],
                                 *(const v8h*)&sAlo[m][kb + 16 + lg * 8]);
                // B fragment: lane<16 -> K {kb..kb+7, kb+16..23}; lane>=16 shifted by 8
                v16h bhi = cat16(*(const v8h*)&sBhi[n][kb + lg * 8],
                                 *(const v8h*)&sBhi[n][kb + 16 + lg * 8]);
                v16h blo = cat16(*(const v8h*)&sBlo[n][kb + lg * 8],
                                 *(const v8h*)&sBlo[n][kb + 16 + lg * 8]);
                acc[cc] = __builtin_amdgcn_wmma_f32_16x16x32_f16(
                    false, ahi, false, bhi, (short)0, acc[cc], false, false);
                acc[cc] = __builtin_amdgcn_wmma_f32_16x16x32_f16(
                    false, ahi, false, blo, (short)0, acc[cc], false, false);
                acc[cc] = __builtin_amdgcn_wmma_f32_16x16x32_f16(
                    false, alo, false, bhi, (short)0, acc[cc], false, false);
            }
        }
    }

    // --- write cost tile: C = max(2 - 2g, 0) ---
#pragma unroll
    for (int cc = 0; cc < 2; ++cc) {
        float* Cb = cost + ((size_t)b * NN + i0 + r * 16) * MM + j0 + (cg + cc) * 16;
#pragma unroll
        for (int e = 0; e < 8; ++e) {
            int Mr = e + lg * 8;
            float g = acc[cc][e];
            Cb[(size_t)Mr * MM + li] = fmaxf(2.0f - 2.0f * g, 0.0f);
        }
    }
}

// ---------------------------------------------------------------------------
// Kernel 3: row update  log_u[b,i] = LOG_A - logsumexp_j(-20*C + log_v[j])
// One wave per row; 16 elements per lane.
// ---------------------------------------------------------------------------
__global__ __launch_bounds__(256) void row_update(const float* __restrict__ cost,
                                                  const float* __restrict__ log_v,
                                                  float* __restrict__ log_u) {
    int wid = threadIdx.x >> 5, lane = threadIdx.x & 31;
    int row = blockIdx.x * 8 + wid;              // b*N + i
    int b   = row >> 9;
    const float* Cr = cost + (size_t)row * MM;
    const float* lv = log_v + b * MM;
    float vals[16];
    float mx = -1e30f;
#pragma unroll
    for (int t = 0; t < 16; ++t) {
        int j = lane + t * 32;
        float v = -INV_EPS * Cr[j] + lv[j];
        vals[t] = v;
        mx = fmaxf(mx, v);
    }
#pragma unroll
    for (int off = 16; off; off >>= 1) mx = fmaxf(mx, __shfl_xor(mx, off, 32));
    float s = 0.0f;
#pragma unroll
    for (int t = 0; t < 16; ++t) s += __expf(vals[t] - mx);
#pragma unroll
    for (int off = 16; off; off >>= 1) s += __shfl_xor(s, off, 32);
    if (lane == 0) log_u[row] = LOG_A - (mx + __logf(s));
}

// ---------------------------------------------------------------------------
// Kernel 4: col update  log_v[b,j] = LOG_B - logsumexp_i(-20*C + log_u[i])
// Block: 64 columns x 4 row-strips, online logsumexp, LDS combine.
// ---------------------------------------------------------------------------
__global__ __launch_bounds__(256) void col_update(const float* __restrict__ cost,
                                                  const float* __restrict__ log_u,
                                                  float* __restrict__ log_v) {
    int b  = blockIdx.x >> 3;
    int j0 = (blockIdx.x & 7) * 64;
    int tx = threadIdx.x & 63, ty = threadIdx.x >> 6;
    int j  = j0 + tx;
    const float* Cb = cost + (size_t)b * NN * MM;
    const float* lu = log_u + b * NN;
    float m = -1e30f, s = 0.0f;
    for (int i = ty; i < NN; i += 4) {
        float v = -INV_EPS * Cb[(size_t)i * MM + j] + lu[i];
        if (v > m) { s = s * __expf(m - v) + 1.0f; m = v; }
        else       { s += __expf(v - m); }
    }
    __shared__ float sm[4][64], ss[4][64];
    sm[ty][tx] = m; ss[ty][tx] = s;
    __syncthreads();
    if (ty == 0) {
#pragma unroll
        for (int k = 1; k < 4; ++k) {
            float m2 = sm[k][tx], s2 = ss[k][tx];
            if (m2 > m) { s = s * __expf(m - m2) + s2; m = m2; }
            else        { s += s2 * __expf(m2 - m); }
        }
        log_v[b * MM + j] = LOG_B - (m + __logf(s));
    }
}

// ---------------------------------------------------------------------------
// Kernel 5: per-block partial of sum(plan * cost); one wave per row.
// ---------------------------------------------------------------------------
__global__ __launch_bounds__(256) void dist_partial(const float* __restrict__ cost,
                                                    const float* __restrict__ log_u,
                                                    const float* __restrict__ log_v,
                                                    float* __restrict__ partials) {
    int wid = threadIdx.x >> 5, lane = threadIdx.x & 31;
    int row = blockIdx.x * 8 + wid;
    int b   = row >> 9;
    const float* Cr = cost + (size_t)row * MM;
    const float* lv = log_v + b * MM;
    float lu = log_u[row];
    float acc = 0.0f;
#pragma unroll
    for (int t = 0; t < 16; ++t) {
        int j = lane + t * 32;
        float c = Cr[j];
        acc += __expf(-INV_EPS * c + lu + lv[j]) * c;
    }
#pragma unroll
    for (int off = 16; off; off >>= 1) acc += __shfl_xor(acc, off, 32);
    __shared__ float sp[8];
    if (lane == 0) sp[wid] = acc;
    __syncthreads();
    if (threadIdx.x == 0) {
        float t = 0.0f;
#pragma unroll
        for (int k = 0; k < 8; ++k) t += sp[k];
        partials[blockIdx.x] = t;
    }
}

// ---------------------------------------------------------------------------
// Kernel 6: final fixed-order reduction of 4096 partials -> mean over batch.
// ---------------------------------------------------------------------------
__global__ __launch_bounds__(256) void final_reduce(const float* __restrict__ partials,
                                                    float* __restrict__ out) {
    float a = 0.0f;
    for (int idx = threadIdx.x; idx < 4096; idx += 256) a += partials[idx];
#pragma unroll
    for (int off = 16; off; off >>= 1) a += __shfl_xor(a, off, 32);
    __shared__ float sp[8];
    int wid = threadIdx.x >> 5, lane = threadIdx.x & 31;
    if (lane == 0) sp[wid] = a;
    __syncthreads();
    if (threadIdx.x == 0) {
        float t = 0.0f;
#pragma unroll
        for (int k = 0; k < 8; ++k) t += sp[k];
        out[0] = t / (float)BB;
    }
}

// ---------------------------------------------------------------------------
extern "C" void kernel_launch(void* const* d_in, const int* in_sizes, int n_in,
                              void* d_out, int out_size, void* d_ws, size_t ws_size,
                              hipStream_t stream) {
    (void)in_sizes; (void)n_in; (void)out_size; (void)ws_size;
    const float* S = (const float*)d_in[0];
    const float* T = (const float*)d_in[1];
    float* out = (float*)d_out;

    float* ws       = (float*)d_ws;
    float* inv_ns   = ws;                       // 32768
    float* inv_nt   = ws + 32768;               // 32768
    float* log_u    = ws + 65536;               // 32768
    float* log_v    = ws + 98304;               // 32768  (contiguous with log_u)
    float* partials = ws + 131072;              // 4096
    float* cost     = ws + 135168;              // 64*512*512 = 16,777,216

    // norms (one wave per row, 8 rows per 256-thread block)
    norms_kernel<<<BB * NN / 8, 256, 0, stream>>>(S, inv_ns);
    norms_kernel<<<BB * MM / 8, 256, 0, stream>>>(T, inv_nt);

    // log_u = log_v = 0  (65536 contiguous floats)
    zero_kernel<<<256, 256, 0, stream>>>(log_u);

    // cost matrix via WMMA: 64 batches x (8x8) 64x64 tiles
    cost_kernel<<<BB * 64, 256, 0, stream>>>(S, T, inv_ns, inv_nt, cost);

    // 80 Sinkhorn iterations (row update then col update, matching scan order)
    for (int it = 0; it < 80; ++it) {
        row_update<<<BB * NN / 8, 256, 0, stream>>>(cost, log_v, log_u);
        col_update<<<BB * 8, 256, 0, stream>>>(cost, log_u, log_v);
    }

    // final distance: sum(plan*cost) per block, then mean over batch
    dist_partial<<<BB * NN / 8, 256, 0, stream>>>(cost, log_u, log_v, partials);
    final_reduce<<<1, 256, 0, stream>>>(partials, out);
}
